// EdgeBlock_45088566673702
// MI455X (gfx1250) — compile-verified, matrix operand order checked by reading, not run on hardware
//
#include <hip/hip_runtime.h>
#include <hip/hip_bf16.h>

typedef __attribute__((ext_vector_type(16))) __bf16 v16bf;
typedef __attribute__((ext_vector_type(8)))  float  v8f;

#define H      128
#define FOURH  512
#define MT     64          // edges per block
#define LDA    520         // LDS A-tile row stride (ushort), padded vs 512
#define LDH    136         // LDS h-tile row stride (ushort), padded vs 128
#define LDO    132         // LDS f32 out-tile row stride (float), padded vs 128
#define LN_EPS 1e-5f

static __device__ __forceinline__ unsigned short f2bf(float f) {
  unsigned int u = __float_as_uint(f);
  u += 0x7FFFu + ((u >> 16) & 1u);     // round-to-nearest-even
  return (unsigned short)(u >> 16);
}

// Load one 16x16x32 bf16 WMMA operand fragment (A-style layout):
// per lane two contiguous 16-byte chunks at +0 and +32 bytes.
static __device__ __forceinline__ v16bf load_frag(const unsigned short* p) {
  union { uint4 q[2]; v16bf v; } u;
  u.q[0] = *(const uint4*)(p);
  u.q[1] = *(const uint4*)(p + 16);
  return u.v;
}

// ---------------------------------------------------------------------------
// Prep: convert x -> bf16, W1/W2 -> transposed bf16 (N-major), g -> bf16.
// ---------------------------------------------------------------------------
__global__ __launch_bounds__(256) void convert_kernel(
    const float* __restrict__ x, const float* __restrict__ W1,
    const float* __restrict__ W2, const float* __restrict__ g,
    unsigned short* __restrict__ xbf, unsigned short* __restrict__ W1t,
    unsigned short* __restrict__ W2t, unsigned short* __restrict__ gbf,
    int NX) {
  int i = blockIdx.x * blockDim.x + threadIdx.x;
  if (i < NX) xbf[i] = f2bf(x[i]);
  if (i < FOURH * H) {                 // W1 is (512,128) row-major: W1[k][n]
    int k = i >> 7, n = i & (H - 1);
    W1t[n * FOURH + k] = f2bf(W1[i]);
  }
  if (i < H * H) {                     // W2 is (128,128): W2[k][n]
    int k = i >> 7, n = i & (H - 1);
    W2t[n * H + k] = f2bf(W2[i]);
  }
  if (i < H) gbf[i] = f2bf(g[i]);
}

// ---------------------------------------------------------------------------
// Fused edge block: gather+concat -> GEMM1+ReLU -> GEMM2 -> LayerNorm.
// One 256-thread block (8 waves) per 64 edges.
// ---------------------------------------------------------------------------
__global__ __launch_bounds__(256) void edge_block_kernel(
    const unsigned short* __restrict__ xbf,
    const int* __restrict__ ei,          // (2,E): [0,E)=src, [E,2E)=tgt
    const float* __restrict__ eattr,
    const unsigned short* __restrict__ gbf,
    const unsigned short* __restrict__ W1t,
    const float* __restrict__ b1,
    const unsigned short* __restrict__ W2t,
    const float* __restrict__ b2,
    const float* __restrict__ lng,
    const float* __restrict__ lnb,
    float* __restrict__ out,
    int E) {
  __shared__ __align__(16) unsigned char smemA[MT * LDA * 2];  // 65 KB
  __shared__ __align__(16) unsigned short sH[MT * LDH];        // 17 KB
  __shared__ float sMu[MT];
  __shared__ float sRs[MT];
  unsigned short* sA = (unsigned short*)smemA;   // bf16 input tile 64x512
  float*          sO = (float*)smemA;            // f32 out tile 64x128 (reuse)

  const int t  = threadIdx.x;
  const int e0 = blockIdx.x * MT;

  // ---- stage edge_input tile [x_tgt | x_src | edge_attr | g] into LDS ----
  {
    const int r   = t >> 2;                 // 0..63 edge row
    const int seg = t & 3;                  // which 128-wide segment
    const int e   = min(e0 + r, E - 1);
    unsigned short* dst = sA + r * LDA + seg * H;
    if (seg == 2) {                         // edge_attr: f32 -> bf16 fused
      const float4* s = (const float4*)(eattr + (size_t)e * H);
      #pragma unroll
      for (int i = 0; i < 32; ++i) {
        float4 v = s[i];
        dst[4 * i + 0] = f2bf(v.x);
        dst[4 * i + 1] = f2bf(v.y);
        dst[4 * i + 2] = f2bf(v.z);
        dst[4 * i + 3] = f2bf(v.w);
      }
    } else {
      const unsigned short* srcp;
      if (seg == 0)      srcp = xbf + (size_t)ei[E + e] * H;   // x_target
      else if (seg == 1) srcp = xbf + (size_t)ei[e] * H;       // x_source
      else               srcp = gbf;                           // global row
      const uint4* s = (const uint4*)srcp;
      uint4*       d = (uint4*)dst;
      #pragma unroll
      for (int i = 0; i < 16; ++i) d[i] = s[i];
    }
  }
  __syncthreads();

  const int lane = t & 31;
  const int wave = t >> 5;
  const int mt   = wave >> 1;       // 0..3: rows [mt*16, mt*16+16)
  const int nh   = wave & 1;        // 0..1: cols [nh*64, nh*64+64)
  const int lm   = lane & 15;
  const int kh   = lane >> 4;       // lane half -> K-offset selector

  // ---------------- GEMM1: h = relu(A(64x512) @ W1 + b1) ----------------
  v8f acc[4] = {};
  for (int kb = 0; kb < FOURH; kb += 32) {
    v16bf a = load_frag(sA + (mt * 16 + lm) * LDA + kb + kh * 8);
    #pragma unroll
    for (int j = 0; j < 4; ++j) {
      const int n = nh * 64 + j * 16 + lm;
      v16bf b = load_frag(W1t + (size_t)n * FOURH + kb + kh * 8);
      acc[j] = __builtin_amdgcn_wmma_f32_16x16x32_bf16(
          false, a, false, b, (short)0, acc[j], false, false);
    }
  }
  #pragma unroll
  for (int j = 0; j < 4; ++j) {
    const int   n    = nh * 64 + j * 16 + lm;
    const float bias = b1[n];
    #pragma unroll
    for (int i = 0; i < 8; ++i) {
      const int m = mt * 16 + kh * 8 + i;     // C layout: lane-half -> M+8
      float v = acc[j][i] + bias;
      v = v > 0.f ? v : 0.f;
      sH[m * LDH + n] = f2bf(v);
    }
  }
  __syncthreads();   // sA reads done; sH complete

  // ---------------- GEMM2: o = h(64x128) @ W2 + b2 ----------------
  v8f acc2[4] = {};
  for (int kb = 0; kb < H; kb += 32) {
    v16bf a = load_frag(sH + (mt * 16 + lm) * LDH + kb + kh * 8);
    #pragma unroll
    for (int j = 0; j < 4; ++j) {
      const int n = nh * 64 + j * 16 + lm;
      v16bf b = load_frag(W2t + (size_t)n * H + kb + kh * 8);
      acc2[j] = __builtin_amdgcn_wmma_f32_16x16x32_bf16(
          false, a, false, b, (short)0, acc2[j], false, false);
    }
  }
  #pragma unroll
  for (int j = 0; j < 4; ++j) {
    const int   n    = nh * 64 + j * 16 + lm;
    const float bias = b2[n];
    #pragma unroll
    for (int i = 0; i < 8; ++i) {
      const int m = mt * 16 + kh * 8 + i;
      sO[m * LDO + n] = acc2[j][i] + bias;    // overlays sA (safe post-barrier)
    }
  }
  __syncthreads();

  // ---------------- LayerNorm stats (one thread per row) ----------------
  if (t < MT) {
    const float* row = sO + t * LDO;
    float s = 0.f, s2 = 0.f;
    #pragma unroll
    for (int i = 0; i < H; ++i) { float v = row[i]; s += v; s2 += v * v; }
    const float mu  = s * (1.0f / H);
    const float var = s2 * (1.0f / H) - mu * mu;
    sMu[t] = mu;
    sRs[t] = rsqrtf(var + LN_EPS);
  }
  __syncthreads();

  // ---------------- normalize + affine, coalesced f32x4 stores ----------
  {
    const int r   = t >> 2;
    const int seg = t & 3;
    const int e   = e0 + r;
    if (e < E) {
      const float  mu  = sMu[r], rs = sRs[r];
      const float* src = sO + r * LDO + seg * 32;
      const float4* gm = (const float4*)(lng + seg * 32);
      const float4* bt = (const float4*)(lnb + seg * 32);
      float4* o = (float4*)(out + (size_t)e * H + seg * 32);
      #pragma unroll
      for (int i = 0; i < 8; ++i) {
        float4 g4 = gm[i], b4 = bt[i], v;
        v.x = (src[4 * i + 0] - mu) * rs * g4.x + b4.x;
        v.y = (src[4 * i + 1] - mu) * rs * g4.y + b4.y;
        v.z = (src[4 * i + 2] - mu) * rs * g4.z + b4.z;
        v.w = (src[4 * i + 3] - mu) * rs * g4.w + b4.w;
        o[i] = v;
      }
    }
  }
}

extern "C" void kernel_launch(void* const* d_in, const int* in_sizes, int n_in,
                              void* d_out, int out_size, void* d_ws, size_t ws_size,
                              hipStream_t stream) {
  const float* x     = (const float*)d_in[0];
  const int*   ei    = (const int*)d_in[1];
  const float* eattr = (const float*)d_in[2];
  const float* g     = (const float*)d_in[3];
  const float* W1    = (const float*)d_in[4];
  const float* b1    = (const float*)d_in[5];
  const float* W2    = (const float*)d_in[6];
  const float* b2    = (const float*)d_in[7];
  const float* lng   = (const float*)d_in[8];
  const float* lnb   = (const float*)d_in[9];

  const int NX = in_sizes[0];        // N*H
  const int E  = in_sizes[1] / 2;    // edge_index is (2,E)

  // workspace layout (bf16 copies), all 16B-aligned
  unsigned short* xbf = (unsigned short*)d_ws;
  unsigned short* W1t = xbf + NX;
  unsigned short* W2t = W1t + FOURH * H;
  unsigned short* gbf = W2t + H * H;

  {
    int total = NX;                  // largest conversion range
    int blocks = (total + 255) / 256;
    convert_kernel<<<blocks, 256, 0, stream>>>(x, W1, W2, g, xbf, W1t, W2t, gbf, NX);
  }
  {
    int blocks = (E + MT - 1) / MT;
    edge_block_kernel<<<blocks, 256, 0, stream>>>(xbf, ei, eattr, gbf, W1t, b1,
                                                  W2t, b2, lng, lnb,
                                                  (float*)d_out, E);
  }
}
